// PropAndPool_11562051960940
// MI455X (gfx1250) — matrix operand order repaired and checked by reading, not compiled, not run on hardware
//
#include <hip/hip_runtime.h>
#include <hip/hip_bf16.h>
#include <math.h>

#define NB     128          // graphs
#define NPER   512          // nodes per graph
#define NN     (NB*NPER)    // 65536 nodes
#define DD     128          // feature dim
#define EE     524288       // edges

typedef __attribute__((ext_vector_type(16))) _Float16 v16h;
typedef __attribute__((ext_vector_type(8)))  _Float16 v8h;
typedef __attribute__((ext_vector_type(8)))  float    v8f;

union AFrag { v16h v; v8h h2[2]; };

// ---------------- utility kernels ----------------

__global__ void fillf_kernel(float* __restrict__ p, float v, long long n) {
    long long i = (long long)blockIdx.x * blockDim.x + threadIdx.x;
    long long stride = (long long)gridDim.x * blockDim.x;
    for (; i < n; i += stride) p[i] = v;
}

__global__ void filli_kernel(int* __restrict__ p, int v, long long n) {
    long long i = (long long)blockIdx.x * blockDim.x + threadIdx.x;
    long long stride = (long long)gridDim.x * blockDim.x;
    for (; i < n; i += stride) p[i] = v;
}

__global__ void cvt_f16_kernel(const float* __restrict__ src, _Float16* __restrict__ dst, int n) {
    int i = blockIdx.x * blockDim.x + threadIdx.x;
    if (i < n) dst[i] = (_Float16)src[i];
}

// ---------------- edge scatter (masked mean-aggregation numerator/denominator) ----------------
// 16 threads per edge, 8 features each, f32 atomics. deg by lane 0.
__global__ void scatter_kernel(const int* __restrict__ ei, const float* __restrict__ h,
                               const int* __restrict__ mask, float* __restrict__ msum,
                               float* __restrict__ deg) {
    long long gid = (long long)blockIdx.x * blockDim.x + threadIdx.x;
    if (gid >= (long long)EE * 16) return;
    int e = (int)(gid >> 4);
    int part = (int)(gid & 15);
    int src = ei[e];
    int dst = ei[EE + e];
    if (mask[src] && mask[dst]) {
        const float* hp = &h[(long long)src * DD + part * 8];
        float* mp = &msum[(long long)dst * DD + part * 8];
#pragma unroll
        for (int i = 0; i < 8; ++i) atomicAdd(&mp[i], hp[i]);
        if (part == 0) atomicAdd(&deg[dst], 1.0f);
    }
}

// ---------------- SAGE conv via WMMA ----------------
// out[n][c] = relu( (msum[n]/max(deg,1)) @ wl.T + bl + h[n] @ wr.T )
// One block (256 thr / 8 waves) per 16-node tile. Wave w owns cols [16w,16w+16).
// out may alias msum (rows are block-private; msum is fully staged to LDS first).
__global__ void __launch_bounds__(256)
sage_wmma_kernel(const float* __restrict__ msum, const float* __restrict__ deg,
                 const float* __restrict__ h,
                 const _Float16* __restrict__ wl16,   // [128][128] row-major f16 (row = out col)
                 const _Float16* __restrict__ wr16,
                 const float* __restrict__ bl,
                 float* __restrict__ out) {
    __shared__ _Float16 sMean[16][136];
    __shared__ _Float16 sH[16][136];
    const int nbase = blockIdx.x * 16;
    const int tid = threadIdx.x;

    for (int i = tid; i < 16 * DD; i += 256) {
        int m = i >> 7, k = i & 127;
        int node = nbase + m;
        float d = deg[node];
        d = d > 1.0f ? d : 1.0f;
        sMean[m][k] = (_Float16)(msum[(long long)node * DD + k] / d);
        sH[m][k]   = (_Float16)(h[(long long)node * DD + k]);
    }
    __syncthreads();

    const int lane  = tid & 31;
    const int w     = tid >> 5;          // wave id: output col tile
    const int mrow  = lane & 15;         // A: M row held by this lane
    const int khalf = lane >> 4;         // 0/1: K-half select
    const int col   = w * 16 + (lane & 15); // B/C: N column held by this lane

    v8f c = {};
#pragma unroll
    for (int kk = 0; kk < 4; ++kk) {     // K = 128 in chunks of 32
        AFrag a;
        int k0 = kk * 32 + khalf * 16;
        a.h2[0] = *(const v8h*)&sMean[mrow][k0];
        a.h2[1] = *(const v8h*)&sMean[mrow][k0 + 8];
        v16h bf = *(const v16h*)&wl16[col * DD + kk * 32 + khalf * 16];
        c = __builtin_amdgcn_wmma_f32_16x16x32_f16(false, a.v, false, bf, (short)0, c, false, false);
    }
#pragma unroll
    for (int kk = 0; kk < 4; ++kk) {
        AFrag a;
        int k0 = kk * 32 + khalf * 16;
        a.h2[0] = *(const v8h*)&sH[mrow][k0];
        a.h2[1] = *(const v8h*)&sH[mrow][k0 + 8];
        v16h bf = *(const v16h*)&wr16[col * DD + kk * 32 + khalf * 16];
        c = __builtin_amdgcn_wmma_f32_16x16x32_f16(false, a.v, false, bf, (short)0, c, false, false);
    }

    float bias = bl[col];
#pragma unroll
    for (int r = 0; r < 8; ++r) {
        int row = khalf * 8 + r;         // C/D layout: lanes>=16 hold M=8..15
        float v = c[r] + bias;
        v = v > 0.0f ? v : 0.0f;
        out[(long long)(nbase + row) * DD + col] = v;
    }
}

// ---------------- TopK pool + readout (one block of 512 thr per graph) ----------------
__global__ void __launch_bounds__(512)
pool_kernel(float* __restrict__ h, int* __restrict__ mask,
            const float* __restrict__ pw, float* __restrict__ reads,
            int ksel, float kf) {
    __shared__ float spw[128];
    __shared__ float sred[128];
    __shared__ float ssc[NPER];
    __shared__ unsigned long long skey[NPER];
    __shared__ unsigned char ssel[NPER];
    const int b = blockIdx.x, tid = threadIdx.x;

    if (tid < 128) spw[tid] = pw[tid];
    __syncthreads();
    if (tid < 128) sred[tid] = spw[tid] * spw[tid];
    __syncthreads();
    for (int s = 64; s > 0; s >>= 1) {
        if (tid < s) sred[tid] += sred[tid + s];
        __syncthreads();
    }
    const float norm = sqrtf(sred[0]) + 1e-16f;

    const int node = b * NPER + tid;
    float sc;
    if (mask[node]) {
        float acc = 0.0f;
        const float* hp = &h[(long long)node * DD];
        for (int d = 0; d < DD; ++d) acc += hp[d] * spw[d];
        sc = acc / norm;
    } else {
        sc = -__builtin_inff();
    }
    ssc[tid] = sc;
    // 64-bit key: descending score, ascending index tie-break, sorted ascending.
    unsigned int u = __float_as_uint(sc);
    u = (u & 0x80000000u) ? ~u : (u | 0x80000000u);   // monotone float->uint (ascending)
    u = ~u;                                           // descending
    skey[tid] = ((unsigned long long)u << 32) | (unsigned int)tid;
    __syncthreads();

    for (int kk = 2; kk <= NPER; kk <<= 1)
        for (int j = kk >> 1; j > 0; j >>= 1) {
            int ixj = tid ^ j;
            if (ixj > tid) {
                bool up = ((tid & kk) == 0);
                unsigned long long a = skey[tid], c2 = skey[ixj];
                if ((a > c2) == up) { skey[tid] = c2; skey[ixj] = a; }
            }
            __syncthreads();
        }

    {
        int li = (int)(skey[tid] & 0xFFFFFFFFull);
        ssel[li] = (tid < ksel) ? (unsigned char)1 : (unsigned char)0;
    }
    __syncthreads();

    const int nm = ssel[tid];
    mask[node] = nm;
    {
        float* hp = &h[(long long)node * DD];
        if (nm) {
            float t = tanhf(ssc[tid]);
            for (int d = 0; d < DD; ++d) hp[d] *= t;
        } else {
            for (int d = 0; d < DD; ++d) hp[d] = 0.0f;
        }
    }
    __syncthreads();   // h + ssel visible block-wide

    if (tid < 128) {
        float gmax = -__builtin_inff(), gsum = 0.0f;
        for (int n2 = 0; n2 < NPER; ++n2) {
            if (ssel[n2]) {
                float v = h[(long long)(b * NPER + n2) * DD + tid];
                gmax = v > gmax ? v : gmax;
                gsum += v;
            }
        }
        reads[b * 256 + tid]       += gmax;
        reads[b * 256 + 128 + tid] += gsum / kf;
    }
}

// ---------------- MLP head: 256 -> 128 -> 64 -> sigmoid(1) ----------------
__global__ void __launch_bounds__(128)
mlp_kernel(const float* __restrict__ reads,
           const float* __restrict__ w1, const float* __restrict__ b1,
           const float* __restrict__ w2, const float* __restrict__ b2,
           const float* __restrict__ w3, const float* __restrict__ b3,
           float* __restrict__ out) {
    __shared__ float r[256], z1[128], z2[64];
    const int b = blockIdx.x, tid = threadIdx.x;
    r[tid]       = reads[b * 256 + tid];
    r[tid + 128] = reads[b * 256 + tid + 128];
    __syncthreads();
    {
        float acc = b1[tid];
        const float* wp = &w1[tid * 256];
        for (int k = 0; k < 256; ++k) acc += wp[k] * r[k];
        z1[tid] = acc > 0.0f ? acc : 0.0f;
    }
    __syncthreads();
    if (tid < 64) {
        float acc = b2[tid];
        const float* wp = &w2[tid * 128];
        for (int k = 0; k < 128; ++k) acc += wp[k] * z1[k];
        z2[tid] = acc > 0.0f ? acc : 0.0f;
    }
    __syncthreads();
    if (tid == 0) {
        float acc = b3[0];
        for (int k = 0; k < 64; ++k) acc += w3[k] * z2[k];
        out[b] = 1.0f / (1.0f + expf(-acc));
    }
}

// ---------------- host launch ----------------

extern "C" void kernel_launch(void* const* d_in, const int* in_sizes, int n_in,
                              void* d_out, int out_size, void* d_ws, size_t ws_size,
                              hipStream_t stream) {
    const float* x  = (const float*)d_in[0];
    const int*   ei = (const int*)d_in[1];
    const float* conv_wl[3] = { (const float*)d_in[2],  (const float*)d_in[6],  (const float*)d_in[10] };
    const float* conv_bl[3] = { (const float*)d_in[3],  (const float*)d_in[7],  (const float*)d_in[11] };
    const float* conv_wr[3] = { (const float*)d_in[4],  (const float*)d_in[8],  (const float*)d_in[12] };
    const float* pool_w[3]  = { (const float*)d_in[5],  (const float*)d_in[9],  (const float*)d_in[13] };
    const float* lin1_w = (const float*)d_in[14];
    const float* lin1_b = (const float*)d_in[15];
    const float* lin2_w = (const float*)d_in[16];
    const float* lin2_b = (const float*)d_in[17];
    const float* lin3_w = (const float*)d_in[18];
    const float* lin3_b = (const float*)d_in[19];
    float* out = (float*)d_out;

    char* ws = (char*)d_ws;
    const size_t SZ_BUF = (size_t)NN * DD * sizeof(float);   // 32 MB
    float* bufA  = (float*)(ws);
    float* bufB  = (float*)(ws + SZ_BUF);
    float* deg   = (float*)(ws + 2 * SZ_BUF);
    int*   mask  = (int*)  (ws + 2 * SZ_BUF + (size_t)NN * 4);
    float* reads = (float*)(ws + 2 * SZ_BUF + (size_t)NN * 8);
    _Float16* wf16 = (_Float16*)(ws + 2 * SZ_BUF + (size_t)NN * 8 + (size_t)NB * 256 * 4);
    // wf16 layout: [layer][wl|wr][128][128]

    const int KS[3] = {410, 328, 263};

    // init: mask = 1, reads = 0, weights -> f16
    filli_kernel<<<256, 256, 0, stream>>>(mask, 1, (long long)NN);
    fillf_kernel<<<64, 256, 0, stream>>>(reads, 0.0f, (long long)NB * 256);
    for (int l = 0; l < 3; ++l) {
        cvt_f16_kernel<<<(DD * DD + 255) / 256, 256, 0, stream>>>(conv_wl[l], wf16 + (size_t)(2 * l) * DD * DD, DD * DD);
        cvt_f16_kernel<<<(DD * DD + 255) / 256, 256, 0, stream>>>(conv_wr[l], wf16 + (size_t)(2 * l + 1) * DD * DD, DD * DD);
    }

    // h0 = x (copy so we can mutate in place during pooling)
    hipMemcpyAsync(bufB, x, SZ_BUF, hipMemcpyDeviceToDevice, stream);

    float* hcur  = bufB;
    float* other = bufA;

    for (int l = 0; l < 3; ++l) {
        float* msum = other;
        fillf_kernel<<<2048, 256, 0, stream>>>(msum, 0.0f, (long long)NN * DD);
        fillf_kernel<<<64, 256, 0, stream>>>(deg, 0.0f, (long long)NN);
        scatter_kernel<<<(int)(((long long)EE * 16 + 255) / 256), 256, 0, stream>>>(ei, hcur, mask, msum, deg);
        sage_wmma_kernel<<<NN / 16, 256, 0, stream>>>(msum, deg, hcur,
                                                      wf16 + (size_t)(2 * l) * DD * DD,
                                                      wf16 + (size_t)(2 * l + 1) * DD * DD,
                                                      conv_bl[l], /*out aliases msum*/ msum);
        pool_kernel<<<NB, NPER, 0, stream>>>(msum, mask, pool_w[l], reads, KS[l], (float)KS[l]);
        other = hcur;
        hcur  = msum;
    }

    mlp_kernel<<<NB, 128, 0, stream>>>(reads, lin1_w, lin1_b, lin2_w, lin2_b, lin3_w, lin3_b, out);
}
